// SMCMR_46952582480108
// MI455X (gfx1250) — matrix operand order repaired and checked by reading, not compiled, not run on hardware
//
#include <hip/hip_runtime.h>
#include <hip/hip_bf16.h>

// Problem constants (match reference)
constexpr int NB = 128;
constexpr int NQ = 256;
constexpr int NS = 512;
constexpr int ND = 1024;

typedef __attribute__((ext_vector_type(16))) __bf16 v16bf;
typedef __attribute__((ext_vector_type(8)))  __bf16 v8bf;
typedef __attribute__((ext_vector_type(8)))  float  v8f;
typedef int vint4 __attribute__((vector_size(16)));   // matches builtin param type

// ---- CDNA5 async global->LDS staging (ASYNCcnt), with portable fallback ----
#ifdef __has_builtin
#  if __has_builtin(__builtin_amdgcn_global_load_async_to_lds_b128)
#    define USE_ASYNC_LDS 1
#  endif
#endif
#ifndef USE_ASYNC_LDS
#  define USE_ASYNC_LDS 0
#endif

#if USE_ASYNC_LDS
#  if defined(__has_builtin) && __has_builtin(__builtin_amdgcn_s_wait_asynccnt)
#    define WAIT_ASYNC(n) __builtin_amdgcn_s_wait_asynccnt(n)
#  else
#    define WAIT_ASYNC(n) asm volatile("s_wait_asynccnt %0" :: "i"(n) : "memory")
#  endif
#else
#  define WAIT_ASYNC(n) ((void)0)
#endif

static __device__ __forceinline__ void lds_copy16(const __bf16* g, __bf16* l) {
#if USE_ASYNC_LDS
    __builtin_amdgcn_global_load_async_to_lds_b128((vint4*)(void*)g,
                                                   (vint4*)(void*)l, 0, 0);
    __builtin_amdgcn_global_load_async_to_lds_b128((vint4*)(void*)(g + 8),
                                                   (vint4*)(void*)(l + 8), 0, 0);
#else
    *(v8bf*)l       = *(const v8bf*)g;
    *(v8bf*)(l + 8) = *(const v8bf*)(g + 8);
#endif
}

static __device__ __forceinline__ __bf16 f2bf(float f) {
    union { float f; unsigned u; } v; v.f = f;
    unsigned short h = (unsigned short)(v.u >> 16);
    __bf16 r;
    __builtin_memcpy(&r, &h, 2);
    return r;
}

static __device__ __forceinline__ v16bf bcat(v8bf lo, v8bf hi) {
    return __builtin_shufflevector(lo, hi, 0, 1, 2, 3, 4, 5, 6, 7,
                                           8, 9, 10, 11, 12, 13, 14, 15);
}

// ---- WMMA fragment loaders (bf16 sources, fully vectorized) ----
// A (16x32 MxK): lane<16 -> row r0+lane, K = [k0+0..15]
//                lane>=16 -> row r0+lane-16, K = [k0+16..31]
static __device__ __forceinline__ v16bf ldA(const __bf16* __restrict__ base,
                                            long long ld, int r0, int k0) {
    int lane = threadIdx.x & 31, lr = lane & 15, half = lane >> 4;
    const __bf16* p = base + (long long)(r0 + lr) * ld + k0 + 16 * half;
    return bcat(*(const v8bf*)p, *(const v8bf*)(p + 8));
}

// B (32x16 KxN): lane owns column n = lane&15; lanes 0-15 cover K=k0..k0+15,
// lanes 16-31 cover K=k0+16..31. Source is N-major [N][K] (pre-transposed).
static __device__ __forceinline__ v16bf ldB(const __bf16* __restrict__ base,
                                            long long ld, int n0, int k0) {
    int lane = threadIdx.x & 31, lr = lane & 15, half = lane >> 4;
    const __bf16* p = base + (long long)(n0 + lr) * ld + k0 + 16 * half;
    return bcat(*(const v8bf*)p, *(const v8bf*)(p + 8));
}

// Same as ldB but from an LDS panel laid out [n][64] bf16 (ld = 64).
static __device__ __forceinline__ v16bf ldB_lds(const __bf16* p0, int n0, int kk) {
    int lane = threadIdx.x & 31, lr = lane & 15, half = lane >> 4;
    const __bf16* p = p0 + (n0 + lr) * 64 + kk + 16 * half;
    return bcat(*(const v8bf*)p, *(const v8bf*)(p + 8));
}

#define WMMA_BF16(A, Bm, C) \
    __builtin_amdgcn_wmma_f32_16x16x32_bf16(false, (A), false, (Bm), (short)0, (C), false, false)

// ============================================================================
// Prep 1: fp32 -> bf16 bulk convert, 8 elems/thread, vectorized.
// ============================================================================
__global__ void k_cvt(const float* __restrict__ src, __bf16* __restrict__ dst, long long n) {
    long long i = ((long long)blockIdx.x * blockDim.x + threadIdx.x) * 8;
    if (i >= n) return;
    float4 a = *(const float4*)(src + i);
    float4 b = *(const float4*)(src + i + 4);
    v8bf o;
    o[0] = f2bf(a.x); o[1] = f2bf(a.y); o[2] = f2bf(a.z); o[3] = f2bf(a.w);
    o[4] = f2bf(b.x); o[5] = f2bf(b.y); o[6] = f2bf(b.z); o[7] = f2bf(b.w);
    *(v8bf*)(dst + i) = o;
}

// ============================================================================
// Prep 2: fp32 [batch][R][C] -> bf16 [batch][C][R] (32x32 tiles via LDS).
// ============================================================================
__global__ void k_tr_bf16(const float* __restrict__ src, __bf16* __restrict__ dst,
                          int R, int C) {
    __shared__ __bf16 tile[32][33];
    long long bi = blockIdx.x;
    int tC = C / 32, tR = R / 32;
    int tc = (int)(bi % tC); bi /= tC;
    int tr = (int)(bi % tR); bi /= tR;
    long long base = bi * (long long)R * C;
    int x  = threadIdx.x & 31;
    int y4 = (threadIdx.x >> 5) * 4;
    const float* sp = src + base + (long long)(tr * 32 + y4) * C + tc * 32 + x;
#pragma unroll
    for (int i = 0; i < 4; ++i) tile[y4 + i][x] = f2bf(sp[(long long)i * C]);
    __syncthreads();
    __bf16* dp = dst + base + (long long)(tc * 32 + y4) * R + tr * 32 + x;
#pragma unroll
    for (int i = 0; i < 4; ++i) dp[(long long)i * R] = tile[x][y4 + i];
}

// ============================================================================
// Kernel 1: scores = leaky_relu(ctx . q^T, 0.1) -> attnT[b,q,s] fp32
//           + sum over q of scores^2 -> sumsq[b,s]
// ============================================================================
__global__ void k_scores(const __bf16* __restrict__ qbf, const __bf16* __restrict__ cbf,
                         float* __restrict__ attnT, float* __restrict__ sumsq) {
    int t = blockIdx.x;
    int sblk = t % (NS / 16); t /= (NS / 16);
    int b = t;
    int wave = threadIdx.x >> 5;

    int s0 = sblk * 16;
    int n0 = wave * 64;

    const __bf16* qb = qbf + (long long)b * NQ * ND;
    const __bf16* cb = cbf + (long long)b * NS * ND;

    v8f acc0 = {}, acc1 = {}, acc2 = {}, acc3 = {};
    for (int k = 0; k < ND; k += 32) {
        __builtin_prefetch(cb + (long long)s0 * ND + k + 128, 0, 0);
        v16bf a  = ldA(cb, ND, s0, k);
        v16bf b0 = ldB(qb, ND, n0 +  0, k);
        v16bf b1 = ldB(qb, ND, n0 + 16, k);
        v16bf b2 = ldB(qb, ND, n0 + 32, k);
        v16bf b3 = ldB(qb, ND, n0 + 48, k);
        acc0 = WMMA_BF16(a, b0, acc0);
        acc1 = WMMA_BF16(a, b1, acc1);
        acc2 = WMMA_BF16(a, b2, acc2);
        acc3 = WMMA_BF16(a, b3, acc3);
    }

    int lane = threadIdx.x & 31, lr = lane & 15, half = lane >> 4;
    v8f accs[4] = {acc0, acc1, acc2, acc3};
#pragma unroll
    for (int j = 0; j < 4; ++j) {
#pragma unroll
        for (int r = 0; r < 8; ++r) {
            int s = s0 + r + 8 * half;
            int q = n0 + 16 * j + lr;
            float x = accs[j][r];
            x = x > 0.f ? x : 0.1f * x;           // leaky_relu(0.1)
            attnT[((long long)b * NQ + q) * NS + s] = x;
            float v = x * x;
#pragma unroll
            for (int off = 1; off < 16; off <<= 1) v += __shfl_xor(v, off, 32);
            if (lr == 0) atomicAdd(&sumsq[b * NS + s], v);
        }
    }
}

// ============================================================================
// Kernel 2: normalize + smoothed softmax over S, emit bf16 probs.
// ============================================================================
__global__ void k_softmax(const float* __restrict__ attnT, const float* __restrict__ sumsq,
                          __bf16* __restrict__ probs, const int* __restrict__ smoothp) {
    int row  = blockIdx.x * 8 + (threadIdx.x >> 5);   // [0, NB*NQ)
    int lane = threadIdx.x & 31;
    int b = row / NQ;
    float sm = (float)(*smoothp);

    const float* src = attnT + (long long)row * NS + lane * 16;
    const float* ss  = sumsq + (long long)b * NS + lane * 16;

    float vals[16];
    float mx = -3.4e38f;
#pragma unroll
    for (int i = 0; i < 16; i += 4) {
        float4 xv = *(const float4*)(src + i);
        float4 sv = *(const float4*)(ss + i);
        vals[i + 0] = xv.x / (sqrtf(sv.x + 1e-8f) + 1e-8f) * sm;
        vals[i + 1] = xv.y / (sqrtf(sv.y + 1e-8f) + 1e-8f) * sm;
        vals[i + 2] = xv.z / (sqrtf(sv.z + 1e-8f) + 1e-8f) * sm;
        vals[i + 3] = xv.w / (sqrtf(sv.w + 1e-8f) + 1e-8f) * sm;
        mx = fmaxf(fmaxf(mx, fmaxf(vals[i], vals[i + 1])), fmaxf(vals[i + 2], vals[i + 3]));
    }
#pragma unroll
    for (int off = 16; off >= 1; off >>= 1) mx = fmaxf(mx, __shfl_xor(mx, off, 32));
    float sum = 0.f;
#pragma unroll
    for (int i = 0; i < 16; ++i) { vals[i] = __expf(vals[i] - mx); sum += vals[i]; }
#pragma unroll
    for (int off = 16; off >= 1; off >>= 1) sum += __shfl_xor(sum, off, 32);
    float inv = 1.f / sum;

    v8bf o0, o1;
#pragma unroll
    for (int i = 0; i < 8; ++i) { o0[i] = f2bf(vals[i] * inv); o1[i] = f2bf(vals[8 + i] * inv); }
    __bf16* dst = probs + (long long)row * NS + lane * 16;
    *(v8bf*)dst = o0;
    *(v8bf*)(dst + 8) = o1;
}

// ============================================================================
// Kernel 3: weighted_ctx (bf16) = probs[b] (QxS) x ctx[b] (SxD), B from ctxT.
// ============================================================================
__global__ void k_wctx(const __bf16* __restrict__ probs, const __bf16* __restrict__ ctxT,
                       __bf16* __restrict__ wc) {
    int t = blockIdx.x;
    int dblk = t % (ND / 256); t /= (ND / 256);
    int qblk = t % (NQ / 16);  t /= (NQ / 16);
    int b = t;
    int wave = threadIdx.x >> 5;

    int m0 = qblk * 16;
    int n0 = dblk * 256 + wave * 64;

    const __bf16* pa = probs + (long long)b * NQ * NS;
    const __bf16* pt = ctxT  + (long long)b * ND * NS;

    v8f acc0 = {}, acc1 = {}, acc2 = {}, acc3 = {};
    for (int k = 0; k < NS; k += 32) {
        v16bf a  = ldA(pa, NS, m0, k);
        v16bf b0 = ldB(pt, NS, n0 +  0, k);
        v16bf b1 = ldB(pt, NS, n0 + 16, k);
        v16bf b2 = ldB(pt, NS, n0 + 32, k);
        v16bf b3 = ldB(pt, NS, n0 + 48, k);
        acc0 = WMMA_BF16(a, b0, acc0);
        acc1 = WMMA_BF16(a, b1, acc1);
        acc2 = WMMA_BF16(a, b2, acc2);
        acc3 = WMMA_BF16(a, b3, acc3);
    }

    int lane = threadIdx.x & 31, lr = lane & 15, half = lane >> 4;
    v8f accs[4] = {acc0, acc1, acc2, acc3};
#pragma unroll
    for (int j = 0; j < 4; ++j)
#pragma unroll
        for (int r = 0; r < 8; ++r) {
            int m = m0 + r + 8 * half;
            wc[((long long)b * NQ + m) * ND + (n0 + 16 * j + lr)] = f2bf(accs[j][r]);
        }
}

// ============================================================================
// Kernel 4: cat = [query, wc] (M=NB*NQ, K=2*ND); dual GEMM vs W_linT / W_gateT.
// Block = 8 waves sharing ONE 32-col B panel (block tile 128m x 32n).
// B panels double-buffered in LDS via GLOBAL_LOAD_ASYNC_TO_LDS_B128 (ASYNCcnt);
// fragments then come from ds_load_b128.  out = q*g + tanh(h)*(1-g).
// ============================================================================
__global__ void __launch_bounds__(256) k_gate(
        const float* __restrict__ query,
        const __bf16* __restrict__ qbf, const __bf16* __restrict__ wcbf,
        const __bf16* __restrict__ WlT, const __bf16* __restrict__ WgT,
        const float* __restrict__ bl, const float* __restrict__ bg,
        float* __restrict__ out) {
    // [buf][mat][n][k]: 2*2*32*64 bf16 = 16 KB
    __shared__ __bf16 lds[2][2][32][64];

    constexpr int NCH = (2 * ND) / 64;          // 32 K-chunks of 64
    int blkN = blockIdx.x % (ND / 32);
    int blkM = blockIdx.x / (ND / 32);
    int wave = threadIdx.x >> 5;
    int m0 = blkM * 128 + wave * 16;
    int n0 = blkN * 32;

    // Staging role: 256 threads move 2 mats x 32 rows x 128B (two b128 each).
    int tid  = threadIdx.x;
    int smat = tid >> 7;             // 0..1  (W_lin / W_gate)
    int srow = (tid >> 2) & 31;      // 0..31 (n within panel)
    int scol = (tid & 3) * 16;       // 0/16/32/48 (bf16 units within chunk)
    const __bf16* gpanel = (smat ? WgT : WlT) + (long long)(n0 + srow) * (2 * ND) + scol;

    // Prime buffer 0
    lds_copy16(gpanel, &lds[0][smat][srow][scol]);

    v8f ah0 = {}, ah1 = {}, ag0 = {}, ag1 = {};
    for (int c = 0; c < NCH; ++c) {
        int kc = c * 64;
        if (c + 1 < NCH) {
            lds_copy16(gpanel + kc + 64, &lds[(c + 1) & 1][smat][srow][scol]);
            WAIT_ASYNC(2);           // retire buffer c (async loads are in-order)
        } else {
            WAIT_ASYNC(0);
        }
        __syncthreads();             // all waves' copies visible

        const __bf16* pl = &lds[c & 1][0][0][0];
        const __bf16* pg = &lds[c & 1][1][0][0];
#pragma unroll
        for (int kk = 0; kk < 64; kk += 32) {
            int k = kc + kk;
            v16bf a = (k < ND) ? ldA(qbf, ND, m0, k) : ldA(wcbf, ND, m0, k - ND);
            v16bf bl0 = ldB_lds(pl,  0, kk);
            v16bf bl1 = ldB_lds(pl, 16, kk);
            v16bf bg0 = ldB_lds(pg,  0, kk);
            v16bf bg1 = ldB_lds(pg, 16, kk);
            ah0 = WMMA_BF16(a, bl0, ah0);
            ah1 = WMMA_BF16(a, bl1, ah1);
            ag0 = WMMA_BF16(a, bg0, ag0);
            ag1 = WMMA_BF16(a, bg1, ag1);
        }
        __syncthreads();             // done reading before next overwrite
    }

    int lane = threadIdx.x & 31, lr = lane & 15, half = lane >> 4;
    v8f hs[2] = {ah0, ah1}, gs[2] = {ag0, ag1};
#pragma unroll
    for (int j = 0; j < 2; ++j)
#pragma unroll
        for (int r = 0; r < 8; ++r) {
            int m = m0 + r + 8 * half;
            int n = n0 + 16 * j + lr;
            float h  = tanhf(hs[j][r] + bl[n]);
            float g  = 1.f / (1.f + __expf(-(gs[j][r] + bg[n])));
            float qv = query[(long long)m * ND + n];
            out[(long long)m * ND + n] = qv * g + h * (1.f - g);
        }
}

// ============================================================================
extern "C" void kernel_launch(void* const* d_in, const int* in_sizes, int n_in,
                              void* d_out, int out_size, void* d_ws, size_t ws_size,
                              hipStream_t stream) {
    const float* query  = (const float*)d_in[0];
    const float* ctx    = (const float*)d_in[1];
    const float* Wl     = (const float*)d_in[2];
    const float* bl     = (const float*)d_in[3];
    const float* Wg     = (const float*)d_in[4];
    const float* bg     = (const float*)d_in[5];
    const int*   smooth = (const int*)d_in[6];
    float* out = (float*)d_out;

    // Workspace layout (256B aligned chunks), total ~512 MB:
    auto align256 = [](size_t x) { return (x + 255) & ~(size_t)255; };
    char* ws = (char*)d_ws;
    size_t off = 0;
    __bf16* qbf  = (__bf16*)(ws + off); off = align256(off + (size_t)NB * NQ * ND * 2);
    __bf16* cbf  = (__bf16*)(ws + off); off = align256(off + (size_t)NB * NS * ND * 2);
    __bf16* ctxT = (__bf16*)(ws + off); off = align256(off + (size_t)NB * ND * NS * 2);
    __bf16* WlT  = (__bf16*)(ws + off); off = align256(off + (size_t)ND * 2 * ND * 2);
    __bf16* WgT  = (__bf16*)(ws + off); off = align256(off + (size_t)ND * 2 * ND * 2);
    float*  attnT = (float*)(ws + off); off = align256(off + (size_t)NB * NQ * NS * 4);
    float*  sumsq = (float*)(ws + off); off = align256(off + (size_t)NB * NS * 4);
    __bf16* probs = (__bf16*)(ws + off); off = align256(off + (size_t)NB * NQ * NS * 2);
    __bf16* wcbf  = (__bf16*)(ws + off); off = align256(off + (size_t)NB * NQ * ND * 2);

    long long nq = (long long)NB * NQ * ND;     // 33.5M
    long long nc = (long long)NB * NS * ND;     // 67.1M

    (void)hipMemsetAsync(sumsq, 0, (size_t)NB * NS * sizeof(float), stream);

    // Precision / layout prep
    k_cvt<<<(int)(nq / 8 / 256), 256, 0, stream>>>(query, qbf, nq);
    k_cvt<<<(int)(nc / 8 / 256), 256, 0, stream>>>(ctx,   cbf, nc);
    k_tr_bf16<<<NB * (NS / 32) * (ND / 32), 256, 0, stream>>>(ctx, ctxT, NS, ND);
    k_tr_bf16<<<(2 * ND / 32) * (ND / 32),  256, 0, stream>>>(Wl,  WlT,  2 * ND, ND);
    k_tr_bf16<<<(2 * ND / 32) * (ND / 32),  256, 0, stream>>>(Wg,  WgT,  2 * ND, ND);

    // Fused attention + gating pipeline (all WMMA)
    k_scores <<<NB * (NS / 16),                     128, 0, stream>>>(qbf, cbf, attnT, sumsq);
    k_softmax<<<(NB * NQ) / 8,                      256, 0, stream>>>(attnT, sumsq, probs, smooth);
    k_wctx   <<<NB * (NQ / 16) * (ND / 256),        128, 0, stream>>>(probs, ctxT, wcbf);
    k_gate   <<<(NB * NQ / 128) * (ND / 32),        256, 0, stream>>>(query, qbf, wcbf, WlT, WgT, bl, bg, out);
}
